// InvolutionBlock_80642305949762
// MI455X (gfx1250) — compile-verified, matrix operand order checked by reading, not run on hardware
//
#include <hip/hip_runtime.h>

// ---------------- problem constants ----------------
#define C_CH   256
#define L_LEN  8192
#define B_N    8
#define KW     7
#define PADW   3
#define CR     64      // C / RED
#define GK     112     // G*K rows of w_span
#define LT     64      // L-tile per workgroup
#define NTHR   128     // 4 waves (wave32)

// padded strides (chosen so col*stride*elemsize is a multiple of 16B)
#define XS_STR  72     // f32, x tile row stride (64 + 6 halo -> 72)
#define HB_STR  72     // bf16, h transposed [col][64]
#define XBT_STR 264    // bf16, x transposed [col][256]
#define ONB_STR 264    // bf16, outn transposed [col][256]
#define OB_STR  260    // f32,  out transposed [col][256]
#define KER_STR 116    // f32,  ker transposed [col][112]

// ---------------- WMMA types ----------------
typedef __attribute__((ext_vector_type(16))) __bf16 v16bf;
typedef __attribute__((ext_vector_type(8)))  __bf16 v8bf;
typedef __attribute__((ext_vector_type(8)))  float  v8f;

// ---------------- LDS layout (bytes) ----------------
// xs    : f32  [256][72]                      73728   (live whole kernel)
// regionA (reused three ways):
//   stages 0-2: wredb 32768 | wspanb 14336 | xbt 33792 | hbt 9216   (90112)
//   stage 3   : obuf_t f32 [64][260]                                (66560)
//   stage 4   : wghb bf16 [256][256]                                (131072)
// ker_t : f32  [64][116]                      29696
// onbt  : bf16 [64][264]                      33792
#define XS_OFF    0
#define REGA_OFF  (C_CH * XS_STR * 4)              // 73728
#define WRED_OFF  (REGA_OFF)
#define WSPAN_OFF (REGA_OFF + 32768)
#define XBT_OFF   (REGA_OFF + 47104)
#define HBT_OFF   (REGA_OFF + 80896)
#define OBUF_OFF  (REGA_OFF)
#define WGH_OFF   (REGA_OFF)
#define KER_OFF   (REGA_OFF + 90112)               // 163840
#define ONB_OFF   (KER_OFF + 29696)                // 193536
#define SMEM_BYTES (ONB_OFF + 33792)               // 227328 B < 320 KB/WGP

// fp32 -> bf16 via hardware convert (v_cvt_pk_bf16_f32 on gfx1250)
static __device__ __forceinline__ __bf16 f2bf(float f) { return (__bf16)f; }

// Unified 16x32 fragment loader from bf16 LDS, two 16-byte ds_load_b128 per lane.
//  A (MxK, row-major weights): src = tile base, stride = row stride.
//  B (KxN, transposed activations [col][k]): src = base + col0*stride + k0.
// Lane layout: m/n = lane&15, half g = lane>>4; K runs { g*8..g*8+7, 16+g*8.. }.
static __device__ __forceinline__ v16bf load_frag(const __bf16* src, int stride, int lane) {
    const int m = lane & 15, g = (lane >> 4) & 1;
    const __bf16* p = src + m * stride + g * 8;
    const v8bf lo = *reinterpret_cast<const v8bf*>(p);
    const v8bf hi = *reinterpret_cast<const v8bf*>(p + 16);
    v16bf f;
#pragma unroll
    for (int i = 0; i < 8; ++i) { f[i] = lo[i]; f[8 + i] = hi[i]; }
    return f;
}

static __device__ __forceinline__ v8f wmma_bf16(v16bf a, v16bf b, v8f c) {
    return __builtin_amdgcn_wmma_f32_16x16x32_bf16(false, a, false, b, (short)0, c,
                                                   false, false);
}

__global__ __launch_bounds__(NTHR)
void InvolutionBlock_80642305949762_kernel(
    const float* __restrict__ x,        const float* __restrict__ w_reduce,
    const float* __restrict__ w_span,   const float* __restrict__ prelu_a,
    const float* __restrict__ ln_gamma, const float* __restrict__ ln_beta,
    const float* __restrict__ w_main,   const float* __restrict__ w_skip,
    float* __restrict__ out)
{
    extern __shared__ char smem[];
    float*  xs     = reinterpret_cast<float*>(smem + XS_OFF);
    __bf16* wredb  = reinterpret_cast<__bf16*>(smem + WRED_OFF);
    __bf16* wspanb = reinterpret_cast<__bf16*>(smem + WSPAN_OFF);
    __bf16* xbt    = reinterpret_cast<__bf16*>(smem + XBT_OFF);
    __bf16* hbt    = reinterpret_cast<__bf16*>(smem + HBT_OFF);
    float*  obuf_t = reinterpret_cast<float*>(smem + OBUF_OFF);
    __bf16* wghb   = reinterpret_cast<__bf16*>(smem + WGH_OFF);
    float*  ker_t  = reinterpret_cast<float*>(smem + KER_OFF);
    __bf16* onbt   = reinterpret_cast<__bf16*>(smem + ONB_OFF);

    const int tid  = threadIdx.x;
    const int lane = tid & 31;
    const int wave = tid >> 5;
    const int n    = lane & 15;         // N position within the wave's 16-col slab
    const int g    = (lane >> 4) & 1;   // lane half
    const int c0   = wave * 16;         // local column base of this wave's slab
    const int l0   = blockIdx.x * LT;   // global L offset of this workgroup
    const int b    = blockIdx.y;

    // ---------------- stage 0: cooperative global -> LDS ----------------
    for (int i = tid; i < C_CH * (LT + 2 * PADW); i += NTHR) {
        const int row = i / (LT + 2 * PADW);
        const int cc  = i % (LT + 2 * PADW);
        const int l   = l0 + cc - PADW;
        float v = 0.0f;
        if (l >= 0 && l < L_LEN)
            v = x[((size_t)b * C_CH + row) * L_LEN + l];
        xs[row * XS_STR + cc] = v;
        if (cc >= PADW && cc < PADW + LT)
            xbt[(cc - PADW) * XBT_STR + row] = f2bf(v);   // transposed bf16 copy
    }
    for (int i = tid; i < CR * C_CH; i += NTHR) wredb[i]  = f2bf(w_reduce[i]);
    for (int i = tid; i < GK * CR;  i += NTHR) wspanb[i] = f2bf(w_span[i]);
    __syncthreads();

    // ---------------- stage 1: h = relu(W_reduce . x)  (64 x 16 per wave) ----------------
    {
        v16bf bxr[8];                                   // x B-fragments, reused x4
#pragma unroll
        for (int kc = 0; kc < C_CH / 32; ++kc)
            bxr[kc] = load_frag(xbt + c0 * XBT_STR + kc * 32, XBT_STR, lane);
#pragma unroll 1
        for (int mt = 0; mt < CR / 16; ++mt) {
            v8f acc = {};
#pragma unroll
            for (int kc = 0; kc < C_CH / 32; ++kc)
                acc = wmma_bf16(load_frag(wredb + (mt * 16) * C_CH + kc * 32, C_CH, lane),
                                bxr[kc], acc);
            v8bf hv;
#pragma unroll
            for (int r = 0; r < 8; ++r) {
                float v = acc[r];
                hv[r] = f2bf(v > 0.0f ? v : 0.0f);
            }
            // rows mt*16+8g .. +7 contiguous -> one 16B ds_store_b128
            *reinterpret_cast<v8bf*>(hbt + (c0 + n) * HB_STR + mt * 16 + 8 * g) = hv;
        }
    }
    // hbt produced & consumed by the same wave (same 16 columns): in-order LDS, no barrier

    // ---------------- stage 2: ker = W_span . h  (112 x 16 per wave) ----------------
    {
        v16bf bhr[2];                                   // h B-fragments, reused x7
#pragma unroll
        for (int kc = 0; kc < CR / 32; ++kc)
            bhr[kc] = load_frag(hbt + c0 * HB_STR + kc * 32, HB_STR, lane);
#pragma unroll 1
        for (int mt = 0; mt < GK / 16; ++mt) {
            v8f acc = {};
#pragma unroll
            for (int kc = 0; kc < CR / 32; ++kc)
                acc = wmma_bf16(load_frag(wspanb + (mt * 16) * CR + kc * 32, CR, lane),
                                bhr[kc], acc);
            float* kp = ker_t + (c0 + n) * KER_STR + mt * 16 + 8 * g;
            *reinterpret_cast<float4*>(kp)     = make_float4(acc[0], acc[1], acc[2], acc[3]);
            *reinterpret_cast<float4*>(kp + 4) = make_float4(acc[4], acc[5], acc[6], acc[7]);
        }
    }
    __syncthreads();   // regionA (weights/xbt/hbt) dead everywhere; obuf_t may overwrite

    // ---------------- stage 3: involution + PReLU + LayerNorm ----------------
    const float ap  = prelu_a[0];
    const int   col = c0 + n;          // local column this lane owns
    const int   cb  = g * (C_CH / 2);  // each lane half covers 128 channels (8 groups)
    const float* kerc = ker_t + col * KER_STR;
    float s = 0.0f, s2 = 0.0f;
#pragma unroll 1
    for (int gi = 0; gi < 8; ++gi) {               // groups owned by this lane half
        const int grp = g * 8 + gi;
        float kk[KW];
#pragma unroll
        for (int k = 0; k < KW; ++k) kk[k] = kerc[grp * KW + k];
#pragma unroll 1
        for (int cc = 0; cc < 16; ++cc) {
            const int c = grp * 16 + cc;
            const float* xp = xs + c * XS_STR + col;   // window l-3 .. l+3
            float acc = 0.0f;
#pragma unroll
            for (int k = 0; k < KW; ++k) acc += kk[k] * xp[k];
            acc = (acc >= 0.0f) ? acc : ap * acc;
            obuf_t[col * OB_STR + c] = acc;
            s += acc; s2 += acc * acc;
        }
    }
    // combine the two lane-halves of this column (wave32 shuffle)
    s  += __shfl_xor(s, 16, 32);
    s2 += __shfl_xor(s2, 16, 32);
    const float mu  = s * (1.0f / C_CH);
    const float var = s2 * (1.0f / C_CH) - mu * mu;
    const float rs  = rsqrtf(var + 1e-5f);
#pragma unroll 1
    for (int c = cb; c < cb + C_CH / 2; c += 8) {
        const float* ob = obuf_t + col * OB_STR + c;
        const float4 v0 = *reinterpret_cast<const float4*>(ob);
        const float4 v1 = *reinterpret_cast<const float4*>(ob + 4);
        const float4 g0 = *reinterpret_cast<const float4*>(ln_gamma + c);
        const float4 g1 = *reinterpret_cast<const float4*>(ln_gamma + c + 4);
        const float4 b0 = *reinterpret_cast<const float4*>(ln_beta + c);
        const float4 b1 = *reinterpret_cast<const float4*>(ln_beta + c + 4);
        v8bf o;
        o[0] = f2bf((v0.x - mu) * rs * g0.x + b0.x);
        o[1] = f2bf((v0.y - mu) * rs * g0.y + b0.y);
        o[2] = f2bf((v0.z - mu) * rs * g0.z + b0.z);
        o[3] = f2bf((v0.w - mu) * rs * g0.w + b0.w);
        o[4] = f2bf((v1.x - mu) * rs * g1.x + b1.x);
        o[5] = f2bf((v1.y - mu) * rs * g1.y + b1.y);
        o[6] = f2bf((v1.z - mu) * rs * g1.z + b1.z);
        o[7] = f2bf((v1.w - mu) * rs * g1.w + b1.w);
        *reinterpret_cast<v8bf*>(onbt + col * ONB_STR + c) = o;   // one b128 store
    }
    __syncthreads();   // onbt complete; obuf_t dead -> regionA reusable as wghb

    // ---------------- stage 4: main = W_main . outn + x ; skip = W_skip . outn ----------------
    const size_t TOT = (size_t)B_N * C_CH * L_LEN;
    v16bf bnr[8];                                       // outn B-fragments, reused x16 x2
#pragma unroll
    for (int kc = 0; kc < C_CH / 32; ++kc)
        bnr[kc] = load_frag(onbt + c0 * ONB_STR + kc * 32, ONB_STR, lane);

#pragma unroll 1
    for (int pass = 0; pass < 2; ++pass) {
        // cooperatively stage this pass's weight matrix as bf16 in LDS
        // (each element loaded/converted once per WG, fully coalesced)
        const float* wsrc = pass ? w_skip : w_main;
        for (int i = tid * 8; i < C_CH * C_CH; i += NTHR * 8) {
            const float4 f0 = *reinterpret_cast<const float4*>(wsrc + i);
            const float4 f1 = *reinterpret_cast<const float4*>(wsrc + i + 4);
            v8bf o;
            o[0] = f2bf(f0.x); o[1] = f2bf(f0.y); o[2] = f2bf(f0.z); o[3] = f2bf(f0.w);
            o[4] = f2bf(f1.x); o[5] = f2bf(f1.y); o[6] = f2bf(f1.z); o[7] = f2bf(f1.w);
            *reinterpret_cast<v8bf*>(wghb + i) = o;
        }
        __syncthreads();

#pragma unroll 1
        for (int mt = 0; mt < C_CH / 16; ++mt) {
            v8f acc = {};
#pragma unroll
            for (int kc = 0; kc < C_CH / 32; ++kc)
                acc = wmma_bf16(load_frag(wghb + (mt * 16) * C_CH + kc * 32, C_CH, lane),
                                bnr[kc], acc);
#pragma unroll
            for (int r = 0; r < 8; ++r) {
                const int row  = mt * 16 + r + 8 * g;
                const int colL = c0 + n;
                const size_t o = ((size_t)b * C_CH + row) * L_LEN + l0 + colL;
                if (pass == 0)
                    out[o] = acc[r] + xs[row * XS_STR + colL + PADW];  // + residual x
                else
                    out[TOT + o] = acc[r];
            }
        }
        __syncthreads();   // all reads of wghb done before next pass overwrites it
    }
}

extern "C" void kernel_launch(void* const* d_in, const int* in_sizes, int n_in,
                              void* d_out, int out_size, void* d_ws, size_t ws_size,
                              hipStream_t stream) {
    (void)in_sizes; (void)n_in; (void)out_size; (void)d_ws; (void)ws_size;
    const float* x        = (const float*)d_in[0];
    const float* w_reduce = (const float*)d_in[1];
    const float* w_span   = (const float*)d_in[2];
    const float* prelu_a  = (const float*)d_in[3];
    const float* ln_gamma = (const float*)d_in[4];
    const float* ln_beta  = (const float*)d_in[5];
    const float* w_main   = (const float*)d_in[6];
    const float* w_skip   = (const float*)d_in[7];
    float* out = (float*)d_out;

    dim3 grid(L_LEN / LT, B_N, 1);   // 128 x 8 workgroups
    dim3 block(NTHR, 1, 1);          // 4 wave32 waves
    InvolutionBlock_80642305949762_kernel<<<grid, block, SMEM_BYTES, stream>>>(
        x, w_reduce, w_span, prelu_a, ln_gamma, ln_beta, w_main, w_skip, out);
}
